// LinearAttention_32959579030331
// MI455X (gfx1250) — matrix operand order repaired and checked by reading, not compiled, not run on hardware
//
#include <hip/hip_runtime.h>
#include <hip/hip_bf16.h>

// ---------------------------------------------------------------------------
// Linear attention, CDNA5 (gfx1250, wave32, WMMA bf16 16x16x32, f32 accum)
//
// Shapes: Lq=Lk=4096, N=4, E=1024, H=16, D=64, T = Lq*N = 16384 tokens.
// ---------------------------------------------------------------------------

#define E_DIM   1024
#define L_SEQ   4096
#define N_BATCH 4
#define H_HEADS 16
#define D_HEAD  64
#define T_TOK   (L_SEQ * N_BATCH)     // 16384
#define NH      (N_BATCH * H_HEADS)   // 64
#define EPS_F   1e-6f

typedef __attribute__((ext_vector_type(16))) __bf16 v16bf;
typedef __attribute__((ext_vector_type(8)))  float  v8f;
typedef __attribute__((ext_vector_type(4)))  unsigned int v4u;

// ---------------------------------------------------------------------------
// WMMA fragment helpers (wave32).
//
// A (16x32 bf16, M x K), row-major source with leading dim lda (elements):
//   lane<16  holds row M=lane,    K = {k0..k0+7, k0+16..k0+23}
//   lane>=16 holds row M=lane-16, K = {k0+8..k0+15, k0+24..k0+31}
// B (32x16 bf16, K x N), column-contiguous source (ldb elements per column):
//   lane<16  holds col N=lane,    K = {k0..k0+7, k0+16..k0+23}
//   lane>=16 holds col N=lane-16, K = {k0+8..k0+15, k0+24..k0+31}
// C/D (16x16 f32): vgpr r, lane -> M = r + 8*(lane>=16), N = lane&15
// ---------------------------------------------------------------------------

__device__ __forceinline__ v16bf load_frag_a_bf16(const __bf16* base, int lda, int lane) {
  const int row  = lane & 15;
  const int half = lane >> 4;
  const __bf16* p = base + (size_t)row * lda + half * 8;
  union { v4u q[2]; v16bf v; } f;
  f.q[0] = *(const v4u*)(p);         // K = half*8 .. half*8+7
  f.q[1] = *(const v4u*)(p + 16);    // K = 16+half*8 .. 16+half*8+7
  return f.v;
}

__device__ __forceinline__ v16bf load_frag_a_f32(const float* base, int lda, int lane) {
  const int row  = lane & 15;
  const int half = lane >> 4;
  const float* p = base + (size_t)row * lda + half * 8;
  v16bf v;
#pragma unroll
  for (int i = 0; i < 8; ++i) v[i]     = (__bf16)p[i];
#pragma unroll
  for (int i = 0; i < 8; ++i) v[8 + i] = (__bf16)p[16 + i];
  return v;
}

__device__ __forceinline__ v16bf load_frag_b_bf16(const __bf16* base, int ldb, int lane) {
  const int col  = lane & 15;
  const int koff = (lane >> 4) * 8;
  const __bf16* p = base + (size_t)col * ldb + koff;
  union { v4u q[2]; v16bf v; } f;
  f.q[0] = *(const v4u*)(p);
  f.q[1] = *(const v4u*)(p + 16);
  return f.v;
}

__device__ __forceinline__ v8f wmma_bf16(v16bf a, v16bf b, v8f c) {
  // 8 args: (neg_a, A, neg_b, B, c_mod, C, reuse_a, reuse_b)
  return __builtin_amdgcn_wmma_f32_16x16x32_bf16(false, a, false, b, (short)0, c,
                                                 false, false);
}

// ---------------------------------------------------------------------------
// Elementwise utility kernels
// ---------------------------------------------------------------------------

__global__ void f32_to_bf16_kernel(const float* __restrict__ src,
                                   __bf16* __restrict__ dst, int n) {
  int i = blockIdx.x * blockDim.x + threadIdx.x;
  if (i < n) dst[i] = (__bf16)src[i];
}

__global__ void zero_f32_kernel(float* __restrict__ p, int n) {
  int i = blockIdx.x * blockDim.x + threadIdx.x;
  if (i < n) p[i] = 0.0f;
}

// ---------------------------------------------------------------------------
// Generic projection GEMM:  C[t, j] = act( sum_i A[t,i] * W[j,i] + b[j] )
//   A: T x 1024 (f32 if A_F32 else bf16), W: 1024x1024 bf16 row-major
//   (rows = output features j, so W row j is the column-contiguous B operand).
// Block: 256 threads = 8 waves. Wave tile: 32(M) x 64(N)  -> block M = 256.
// Per K-step: 2 A frags + 4 B frags -> 8 WMMAs (B reuse doubled vs M=16).
// OUT_MODE 0: bf16 token-major [t*1024 + j]
// OUT_MODE 1: bf16 transposed  [((n*16+h)*64 + d) * L + l]  (t=l*4+n, j=h*64+d)
// OUT_MODE 2: f32  token-major [t*1024 + j]  (final output)
// ACT 1: elu(x)+1  (== exp(x) for x<0, x+1 for x>=0)
// ---------------------------------------------------------------------------

template <int ACT, int OUT_MODE, int A_F32>
__global__ void __launch_bounds__(256) gemm_proj_kernel(
    const void* __restrict__ Aptr, const __bf16* __restrict__ W,
    const float* __restrict__ bias, __bf16* __restrict__ out_bf,
    float* __restrict__ out_f32) {
  const int lane = threadIdx.x & 31;
  const int wave = threadIdx.x >> 5;
  const int m0 = (blockIdx.y * 8 + wave) * 32;  // token tile (2 x 16 rows)
  const int n0 = blockIdx.x * 64;               // output-feature tile

  v8f acc[2][4];
#pragma unroll
  for (int s = 0; s < 2; ++s)
#pragma unroll
    for (int t = 0; t < 4; ++t) acc[s][t] = v8f{0.f};

  for (int k0 = 0; k0 < E_DIM; k0 += 32) {
    v16bf a[2];
    if (A_F32) {
      const float* Abase = (const float*)Aptr + (size_t)m0 * E_DIM + k0;
      if (k0 + 32 < E_DIM) __builtin_prefetch(Abase + 32, 0, 1);  // global_prefetch_b8
#pragma unroll
      for (int s = 0; s < 2; ++s)
        a[s] = load_frag_a_f32(Abase + (size_t)(s * 16) * E_DIM, E_DIM, lane);
    } else {
      const __bf16* Abase = (const __bf16*)Aptr + (size_t)m0 * E_DIM + k0;
      if (k0 + 32 < E_DIM) __builtin_prefetch(Abase + 32, 0, 1);
#pragma unroll
      for (int s = 0; s < 2; ++s)
        a[s] = load_frag_a_bf16(Abase + (size_t)(s * 16) * E_DIM, E_DIM, lane);
    }
#pragma unroll
    for (int t = 0; t < 4; ++t) {
      v16bf b = load_frag_b_bf16(W + (size_t)(n0 + 16 * t) * E_DIM + k0, E_DIM, lane);
#pragma unroll
      for (int s = 0; s < 2; ++s) acc[s][t] = wmma_bf16(a[s], b, acc[s][t]);
    }
  }

  // Epilogue: bias + activation + store per C/D layout.
  const int col = lane & 15;
  const int hi  = lane >> 4;
#pragma unroll
  for (int t = 0; t < 4; ++t) {
    const int j = n0 + 16 * t + col;
    const float bj = bias[j];
#pragma unroll
    for (int s = 0; s < 2; ++s) {
#pragma unroll
      for (int r = 0; r < 8; ++r) {
        const int m = m0 + s * 16 + r + hi * 8;  // token index t
        float x = acc[s][t][r] + bj;
        if (ACT) x = (x >= 0.0f) ? (x + 1.0f) : __expf(x);
        if (OUT_MODE == 0) {
          out_bf[(size_t)m * E_DIM + j] = (__bf16)x;
        } else if (OUT_MODE == 1) {
          const int l = m >> 2, nb = m & 3;      // t = l*N + n, N=4
          const int h = j >> 6, d = j & 63;      // j = h*64 + d
          out_bf[((size_t)((nb * H_HEADS + h) * D_HEAD + d)) * L_SEQ + l] = (__bf16)x;
        } else {
          out_f32[(size_t)m * E_DIM + j] = x;
        }
      }
    }
  }
}

// ---------------------------------------------------------------------------
// k_sum[nh, d] = sum_l kfT[nh, d, l]   (kfT is (NH, 64, L), K-contiguous)
// ---------------------------------------------------------------------------

__global__ void ksum_kernel(const __bf16* __restrict__ kfT,
                            float* __restrict__ ksum) {
  const int nh = blockIdx.x;
  const int d  = threadIdx.x;  // blockDim = 64
  const __bf16* p = kfT + ((size_t)nh * D_HEAD + d) * L_SEQ;
  float s = 0.0f;
  for (int l = 0; l < L_SEQ; l += 8) {
    union { v4u q; __bf16 h[8]; } u;
    u.q = *(const v4u*)(p + l);
#pragma unroll
    for (int i = 0; i < 8; ++i) s += (float)u.h[i];
  }
  ksum[nh * D_HEAD + d] = s;
}

// ---------------------------------------------------------------------------
// kv[nh][e,d] += sum_l kfT[nh,d,l] * vpT[nh,e,l]
//   GEMM per head: M=d=64, N=e=64, K=l=4096, K split over gridDim.y chunks,
//   f32 atomic reduction into kvT (e-major so stage-3 B is K-contiguous).
// Block: 128 threads = 4 waves; each wave owns a K sub-range of 128.
// ---------------------------------------------------------------------------

__global__ void __launch_bounds__(128) kv_kernel(
    const __bf16* __restrict__ kfT, const __bf16* __restrict__ vpT,
    float* __restrict__ kvT) {
  const int nh   = blockIdx.x;
  const int lane = threadIdx.x & 31;
  const int wave = threadIdx.x >> 5;
  const int kb   = blockIdx.y * 512 + wave * 128;

  const __bf16* abase = kfT + (size_t)nh * D_HEAD * L_SEQ;
  const __bf16* bbase = vpT + (size_t)nh * D_HEAD * L_SEQ;

  v8f acc[4][4];
#pragma unroll
  for (int i = 0; i < 4; ++i)
#pragma unroll
    for (int j = 0; j < 4; ++j) acc[i][j] = v8f{0.f};

  for (int k0 = kb; k0 < kb + 128; k0 += 32) {
    v16bf a[4];
#pragma unroll
    for (int di = 0; di < 4; ++di)
      a[di] = load_frag_a_bf16(abase + (size_t)(di * 16) * L_SEQ + k0, L_SEQ, lane);
#pragma unroll
    for (int ei = 0; ei < 4; ++ei) {
      v16bf b = load_frag_b_bf16(bbase + (size_t)(ei * 16) * L_SEQ + k0, L_SEQ, lane);
#pragma unroll
      for (int di = 0; di < 4; ++di) acc[di][ei] = wmma_bf16(a[di], b, acc[di][ei]);
    }
  }

  const int col = lane & 15;
  const int hi  = lane >> 4;
  float* kvbase = kvT + (size_t)nh * D_HEAD * D_HEAD;
#pragma unroll
  for (int di = 0; di < 4; ++di)
#pragma unroll
    for (int ei = 0; ei < 4; ++ei)
#pragma unroll
      for (int r = 0; r < 8; ++r) {
        const int d = di * 16 + r + hi * 8;
        const int e = ei * 16 + col;
        atomicAdd(&kvbase[e * D_HEAD + d], acc[di][ei][r]);
      }
}

// ---------------------------------------------------------------------------
// Stage 3: attn[l, n, h*64+e] = z(l) * sum_d qf[l,n,h*64+d] * kv[nh][e,d]
//   z(l) = 1 / (sum_d qf[l,d] * ksum[nh,d] + eps)
// Block: 256 = 8 waves, each wave: M=16 rows of l, N=64 cols e, K=64.
// ---------------------------------------------------------------------------

__global__ void __launch_bounds__(256) stage3_kernel(
    const __bf16* __restrict__ qf, const __bf16* __restrict__ kv_bf,
    const float* __restrict__ ksum, __bf16* __restrict__ attn) {
  const int nh   = blockIdx.x;
  const int h    = nh & (H_HEADS - 1);
  const int nb   = nh >> 4;
  const int lane = threadIdx.x & 31;
  const int wave = threadIdx.x >> 5;
  const int l0   = blockIdx.y * 128 + wave * 16;

  const int LDA = N_BATCH * E_DIM;  // 4096 elements between consecutive l
  const __bf16* Abase = qf + (size_t)nb * E_DIM + h * D_HEAD + (size_t)l0 * LDA;
  const __bf16* Bbase = kv_bf + (size_t)nh * D_HEAD * D_HEAD;

  v8f acc[4] = {v8f{0.f}, v8f{0.f}, v8f{0.f}, v8f{0.f}};
#pragma unroll
  for (int k0 = 0; k0 < D_HEAD; k0 += 32) {
    v16bf a = load_frag_a_bf16(Abase + k0, LDA, lane);
#pragma unroll
    for (int ei = 0; ei < 4; ++ei) {
      v16bf b = load_frag_b_bf16(Bbase + (size_t)(ei * 16) * D_HEAD + k0, D_HEAD, lane);
      acc[ei] = wmma_bf16(a, b, acc[ei]);
    }
  }

  // z per row (lanes 0..15 each handle one row of this wave's M-tile).
  __shared__ float zsh[8][16];
  if (lane < 16) {
    const __bf16* qrow = qf + (size_t)(l0 + lane) * LDA + (size_t)nb * E_DIM + h * D_HEAD;
    const float* ks = ksum + nh * D_HEAD;
    float s = 0.0f;
#pragma unroll
    for (int d = 0; d < D_HEAD; ++d) s += (float)qrow[d] * ks[d];
    zsh[wave][lane] = 1.0f / (s + EPS_F);
  }
  __syncthreads();

  const int col = lane & 15;
  const int hi  = lane >> 4;
#pragma unroll
  for (int ei = 0; ei < 4; ++ei)
#pragma unroll
    for (int r = 0; r < 8; ++r) {
      const int m = r + hi * 8;
      const int l = l0 + m;
      const int e = ei * 16 + col;
      const float x = acc[ei][r] * zsh[wave][m];
      attn[(size_t)(l * N_BATCH + nb) * E_DIM + h * D_HEAD + e] = (__bf16)x;
    }
}

// ---------------------------------------------------------------------------
// Host launch
// ---------------------------------------------------------------------------

extern "C" void kernel_launch(void* const* d_in, const int* in_sizes, int n_in,
                              void* d_out, int out_size, void* d_ws, size_t ws_size,
                              hipStream_t stream) {
  (void)in_sizes; (void)n_in; (void)out_size; (void)ws_size;
  const float* q  = (const float*)d_in[0];
  const float* k  = (const float*)d_in[1];
  const float* v  = (const float*)d_in[2];
  const float* Wq = (const float*)d_in[3];
  const float* bq = (const float*)d_in[4];
  const float* Wk = (const float*)d_in[5];
  const float* bk = (const float*)d_in[6];
  const float* Wv = (const float*)d_in[7];
  const float* bv = (const float*)d_in[8];
  const float* Wo = (const float*)d_in[9];
  const float* bo = (const float*)d_in[10];
  float* out = (float*)d_out;

  // Workspace layout (bytes)
  char* ws = (char*)d_ws;
  const size_t MB = 1024ull * 1024ull;
  __bf16* Wq_bf = (__bf16*)(ws + 0 * MB);
  __bf16* Wk_bf = (__bf16*)(ws + 2 * MB);
  __bf16* Wv_bf = (__bf16*)(ws + 4 * MB);
  __bf16* Wo_bf = (__bf16*)(ws + 6 * MB);
  __bf16* qf    = (__bf16*)(ws + 8 * MB);                    // T x E, token-major
  __bf16* kfT   = (__bf16*)(ws + 40 * MB);                   // (NH,64,L)
  __bf16* vpT   = (__bf16*)(ws + 72 * MB);                   // (NH,64,L)
  __bf16* attn  = (__bf16*)(ws + 104 * MB);                  // T x E, token-major
  float*  kvT   = (float*)(ws + 136 * MB);                   // (NH,64,64) e-major, f32
  __bf16* kv_bf = (__bf16*)(ws + 138 * MB);                  // (NH,64,64) bf16
  float*  ksum  = (float*)(ws + 139 * MB);                   // (NH,64) f32

  const int WE = E_DIM * E_DIM;  // 1 M elements per weight matrix
  const dim3 cvt_grid((WE + 255) / 256);
  f32_to_bf16_kernel<<<cvt_grid, 256, 0, stream>>>(Wq, Wq_bf, WE);
  f32_to_bf16_kernel<<<cvt_grid, 256, 0, stream>>>(Wk, Wk_bf, WE);
  f32_to_bf16_kernel<<<cvt_grid, 256, 0, stream>>>(Wv, Wv_bf, WE);
  f32_to_bf16_kernel<<<cvt_grid, 256, 0, stream>>>(Wo, Wo_bf, WE);

  const int KVN = NH * D_HEAD * D_HEAD;  // 262144
  zero_f32_kernel<<<(KVN + 255) / 256, 256, 0, stream>>>(kvT, KVN);

  // Projections: grid = (E/64 N-tiles, T/256 M-tiles), 256 thr = 8 waves,
  // each wave 32x64 output.
  const dim3 pgrid(E_DIM / 64, T_TOK / 256);
  gemm_proj_kernel<1, 0, 1><<<pgrid, 256, 0, stream>>>(q, Wq_bf, bq, qf, nullptr);
  gemm_proj_kernel<1, 1, 1><<<pgrid, 256, 0, stream>>>(k, Wk_bf, bk, kfT, nullptr);
  gemm_proj_kernel<0, 1, 1><<<pgrid, 256, 0, stream>>>(v, Wv_bf, bv, vpT, nullptr);

  ksum_kernel<<<NH, 64, 0, stream>>>(kfT, ksum);

  kv_kernel<<<dim3(NH, L_SEQ / 512), 128, 0, stream>>>(kfT, vpT, kvT);

  f32_to_bf16_kernel<<<(KVN + 255) / 256, 256, 0, stream>>>(kvT, kv_bf, KVN);

  stage3_kernel<<<dim3(NH, L_SEQ / 128), 256, 0, stream>>>(qf, kv_bf, ksum, attn);

  gemm_proj_kernel<0, 2, 0><<<pgrid, 256, 0, stream>>>(attn, Wo_bf, bo, nullptr, out);
}
